// DCTLAVISBlip_54786602828026
// MI455X (gfx1250) — matrix operand order repaired and checked by reading, not compiled, not run on hardware
//
#include <hip/hip_runtime.h>

// ---------------------------------------------------------------------------
// DCT -> threshold/truncate -> IDCT pipeline for MI455X (gfx1250, wave32).
// GEMMs: v_wmma_f32_16x16x32_f16 (f16 A/B, f32 accumulate), double-buffered
// LDS software pipeline, async global->LDS copy (ASYNCcnt) for the A tiles.
// L (truncation length) is computed on-device; every dependent kernel is
// launched at max size and self-guards on L (graph-capture safe).
// ---------------------------------------------------------------------------

#define B_  64
#define T_  576
#define C_  1024
#define QIDX 460          // 0.8 * (576 - 1), exact integer -> sorted[460]
#define PI_F 3.14159265358979323846f

typedef __attribute__((ext_vector_type(16))) _Float16 v16h;
typedef __attribute__((ext_vector_type(8)))  _Float16 v8h;
typedef __attribute__((ext_vector_type(4)))  _Float16 v4h;
typedef __attribute__((ext_vector_type(8)))  float    v8f;

// Workspace layout (bytes)
#define WS_MEANSUM 0                       // f32[576]
#define WS_THR     2304                    // f32
#define WS_L       2308                    // i32
#define WS_MF16    4096                    // f16[576*576]  DCT matrix M
#define WS_MIT     667648                  // f16[576*576]  Mi^T (IDCT, L-dep)
#define WS_XDCT    1331200                 // f32[64*576*1024] intermediate

__device__ __forceinline__ void wait_asynccnt0() {
#if __has_builtin(__builtin_amdgcn_s_wait_asynccnt)
    __builtin_amdgcn_s_wait_asynccnt(0);
#else
    asm volatile("s_wait_asynccnt 0x0" ::: "memory");
#endif
}

// ---------------------------------------------------------------------------
__global__ __launch_bounds__(1024)
void dct_init(float* __restrict__ meansum, int* __restrict__ Lp,
              float* __restrict__ thr) {
    int i = threadIdx.x;
    if (i < T_) meansum[i] = 0.0f;
    if (i == 0) { Lp[0] = 1; thr[0] = 0.0f; }
}

// M[k][t] = s_k * cos(pi*(2t+1)*k / (2T)),  s_0 = sqrt(1/T), s_k = sqrt(2/T)
__global__ __launch_bounds__(256)
void dct_gen_M(_Float16* __restrict__ Mf16) {
    int idx = blockIdx.x * 256 + threadIdx.x;
    if (idx >= T_ * T_) return;
    int k = idx / T_, t = idx % T_;
    float s = sqrtf((k == 0 ? 1.0f : 2.0f) / (float)T_);
    float a = PI_F * (float)((2 * t + 1) * k) / (2.0f * (float)T_);
    Mf16[idx] = (_Float16)(s * cosf(a));
}

// MiT[k][l] = Mi[l][k] = s_l * cos(pi*(2k+1)*l / (2L)); stored stride T_
__global__ __launch_bounds__(256)
void dct_gen_MiT(_Float16* __restrict__ MiT, const int* __restrict__ Lp) {
    int L = Lp[0];
    int idx = blockIdx.x * 256 + threadIdx.x;
    if (idx >= T_ * T_) return;
    int k = idx / T_, l = idx % T_;
    if (k >= L || l >= L) return;
    float s = sqrtf((l == 0 ? 1.0f : 2.0f) / (float)L);
    float a = PI_F * (float)((2 * k + 1) * l) / (2.0f * (float)L);
    MiT[(size_t)k * T_ + l] = (_Float16)(s * cosf(a));
}

// ---------------------------------------------------------------------------
// Batched GEMM, tile 64(k) x 128(c), BK = 32, 8 waves (2x4, each 32x32 via
// four 16x16x32 f16 WMMAs), ping/pong LDS pipeline.
// MODE 0: x_dct = M @ x[b] (+ fused mean-sum).  MODE 1: state = MiT @ x_dct.
// ---------------------------------------------------------------------------
template<int MODE>
__global__ __launch_bounds__(256)
void dct_gemm(const _Float16* __restrict__ Amat,   // [*][T_] f16, stride T_
              const float*    __restrict__ Bmat,   // [b][T_][C_] f32
              float*          __restrict__ xdctOut,// MODE 0 output
              void*           __restrict__ dout,   // MODE 1 output (f16)
              float*          __restrict__ meansum,
              const int*      __restrict__ Lp)
{
    const int L    = (MODE == 0) ? T_ : Lp[0];
    const int Kdim = L;
    const int kTile = blockIdx.y * 64;
    if (MODE == 1 && kTile >= L) return;          // uniform early-exit
    const int cTile = blockIdx.x * 128;
    const int b     = blockIdx.z;

    __shared__ _Float16 As[2][64 * 32];           // [k][t], stride 32 halves
    __shared__ _Float16 Bs[2][128 * 40];          // [c][t], stride 40 halves

    const int tid    = threadIdx.x;
    const int lane   = tid & 31;
    const int laneLo = lane & 15;
    const int laneHi = lane >> 4;                 // 0 or 1
    const int waveId = tid >> 5;
    const int waveM  = waveId & 1;                // 2 row-groups of 32
    const int waveN  = waveId >> 1;               // 4 col-groups of 32

    v8f acc[2][2] = {};
    const float* Bbase = Bmat + (size_t)b * T_ * C_;
    const int nIter = (Kdim + 31) >> 5;

    // ---- staging helpers -------------------------------------------------
    // A tile 64x32 f16: async global->LDS copy (one b128 per thread).
    auto stageA = [&](int tb, int bufIdx) {
        const int row = tid >> 2;                 // 0..63
        const int cq  = (tid & 3) << 3;           // 0,8,16,24
        const _Float16* src = Amat + (size_t)(kTile + row) * T_ + tb + cq;
        _Float16* dst = &As[bufIdx][row * 32 + cq];
        if (tb + 32 <= Kdim) {
            unsigned      ldsDst = (unsigned)(size_t)dst;       // LDS = addr[31:0]
            unsigned long long g = (unsigned long long)(size_t)src;
            asm volatile("global_load_async_to_lds_b128 %0, %1, off"
                         :: "v"(ldsDst), "v"(g) : "memory");
        } else {
            #pragma unroll
            for (int i = 0; i < 8; ++i)
                dst[i] = (tb + cq + i < Kdim) ? src[i] : (_Float16)0.0f;
        }
    };
    // B tile 32(t)x128(c) fp32 -> f16 transposed: thread owns 4t x 4c.
    auto loadB = [&](int tb, float4* f) {
        const int r0 = (tid >> 5) << 2;           // 0,4,...,28
        const int c0 = (tid & 31) << 2;           // 0..124
        #pragma unroll
        for (int j = 0; j < 4; ++j) {
            if (tb + r0 + j < Kdim)
                f[j] = *(const float4*)(Bbase + (size_t)(tb + r0 + j) * C_ + cTile + c0);
            else
                f[j] = make_float4(0.f, 0.f, 0.f, 0.f);
        }
    };
    auto storeB = [&](int bufIdx, const float4* f) {
        const int r0 = (tid >> 5) << 2;
        const int c0 = (tid & 31) << 2;
        #pragma unroll
        for (int i = 0; i < 4; ++i) {             // 4 x ds_store_b64
            v4h h;
            h[0] = (_Float16)((const float*)&f[0])[i];
            h[1] = (_Float16)((const float*)&f[1])[i];
            h[2] = (_Float16)((const float*)&f[2])[i];
            h[3] = (_Float16)((const float*)&f[3])[i];
            *(v4h*)(&Bs[bufIdx][(c0 + i) * 40 + r0]) = h;
        }
    };

    // ---- prologue: stage tile 0 into buffer 0 ----------------------------
    {
        float4 fB[4];
        stageA(0, 0);
        loadB(0, fB);
        storeB(0, fB);
        wait_asynccnt0();
        __syncthreads();
    }

    // ---- main pipelined loop --------------------------------------------
    int cur = 0;
    for (int it = 0; it < nIter; ++it) {
        const int nxt = cur ^ 1;
        const bool hasNext = (it + 1) < nIter;
        float4 fB[4];
        if (hasNext) {                            // issue next-tile traffic early
            const int tb = (it + 1) << 5;
            stageA(tb, nxt);
            loadB(tb, fB);
            __builtin_prefetch(Bbase + (size_t)(tb + 32 + ((tid >> 5) << 2)) * C_ +
                               cTile + ((tid & 31) << 2), 0, 0);
        }

        // ---- fragments per ISA VGPR layouts, from buffer `cur` ----
        v16h afrag[2], bfrag[2];
        #pragma unroll
        for (int fr = 0; fr < 2; ++fr) {
            // A 16x32: lane row = laneLo; halves K 0-7/16-23 (laneHi=0),
            // K 8-15/24-31 (laneHi=1)
            const int row = waveM * 32 + fr * 16 + laneLo;
            const _Float16* p = &As[cur][row * 32 + laneHi * 8];
            v8h lo = *(const v8h*)p;
            v8h hi = *(const v8h*)(p + 16);
            #pragma unroll
            for (int i = 0; i < 8; ++i) { afrag[fr][i] = lo[i]; afrag[fr][i + 8] = hi[i]; }
        }
        #pragma unroll
        for (int fc = 0; fc < 2; ++fc) {
            // B 32x16: lane col = laneLo; K 0-7/16-23 (laneHi=0), K 8-15/24-31 (laneHi=1)
            const int col = waveN * 32 + fc * 16 + laneLo;
            const _Float16* p = &Bs[cur][col * 40 + laneHi * 8];
            v8h lo = *(const v8h*)p;
            v8h hi = *(const v8h*)(p + 16);
            #pragma unroll
            for (int i = 0; i < 8; ++i) { bfrag[fc][i] = lo[i]; bfrag[fc][i + 8] = hi[i]; }
        }

        #pragma unroll
        for (int fr = 0; fr < 2; ++fr)
            #pragma unroll
            for (int fc = 0; fc < 2; ++fc)
                acc[fr][fc] = __builtin_amdgcn_wmma_f32_16x16x32_f16(
                    false, afrag[fr], false, bfrag[fc],
                    (short)0, acc[fr][fc], false, false);

        if (hasNext) storeB(nxt, fB);             // convert+pack after WMMA issue
        wait_asynccnt0();                         // A async copy done
        __syncthreads();
        cur = nxt;
    }

    // ---- epilogue ----
    // D layout: VGPR j -> row M=j (lanes 0-15) / M=j+8 (lanes 16-31), col N=laneLo
    if (MODE == 0) {
        float* xd = xdctOut + (size_t)b * T_ * C_;
        #pragma unroll
        for (int fr = 0; fr < 2; ++fr) {
            float rowSum[8];
            #pragma unroll
            for (int j = 0; j < 8; ++j) rowSum[j] = acc[fr][0][j] + acc[fr][1][j];
            #pragma unroll
            for (int fc = 0; fc < 2; ++fc) {
                const int col = cTile + waveN * 32 + fc * 16 + laneLo;
                #pragma unroll
                for (int j = 0; j < 8; ++j) {
                    const int row = kTile + waveM * 32 + fr * 16 + laneHi * 8 + j;
                    xd[(size_t)row * C_ + col] = acc[fr][fc][j];
                }
            }
            // reduce across the 16 lanes of each half-wave (cols)
            #pragma unroll
            for (int m = 1; m < 16; m <<= 1)
                #pragma unroll
                for (int j = 0; j < 8; ++j)
                    rowSum[j] += __shfl_xor(rowSum[j], m, 32);
            if (laneLo == 0) {
                #pragma unroll
                for (int j = 0; j < 8; ++j) {
                    const int row = kTile + waveM * 32 + fr * 16 + laneHi * 8 + j;
                    atomicAdd(&meansum[row], rowSum[j]);
                }
            }
        }
    } else {
        _Float16* hOut = (_Float16*)dout;   // state: [b][L][C] fp16
        #pragma unroll
        for (int fr = 0; fr < 2; ++fr)
            #pragma unroll
            for (int fc = 0; fc < 2; ++fc) {
                const int col = cTile + waveN * 32 + fc * 16 + laneLo;
                #pragma unroll
                for (int j = 0; j < 8; ++j) {
                    const int row = kTile + waveM * 32 + fr * 16 + laneHi * 8 + j;
                    if (row < L)
                        hOut[((size_t)b * L + row) * C_ + col] = (_Float16)acc[fr][fc][j];
                }
            }
    }
}

// ---------------------------------------------------------------------------
// Mean -> abs -> exact order statistic (rank 460) -> last index above -> L
// ---------------------------------------------------------------------------
__global__ __launch_bounds__(576)
void dct_stats(const float* __restrict__ meansum, float* __restrict__ thrOut,
               int* __restrict__ Lp)
{
    __shared__ float v[T_];
    __shared__ float thr;
    __shared__ int lastIdx;
    const int i = threadIdx.x;
    v[i] = fabsf(meansum[i] * (1.0f / (float)(B_ * C_)));
    if (i == 0) lastIdx = 0;
    __syncthreads();

    const float vi = v[i];
    int less = 0, eq = 0;
    for (int j = 0; j < T_; ++j) {
        float vj = v[j];
        less += (vj < vi);
        eq   += (vj == vi);
    }
    if (less <= QIDX && QIDX < less + eq) thr = vi;   // all writers agree
    __syncthreads();
    if (vi > thr) atomicMax(&lastIdx, i);
    __syncthreads();
    if (i == 0) {
        int L = lastIdx;                // slice [:last_index] -> exclusive
        if (L < 1) L = 1;
        Lp[0] = L;
        thrOut[0] = thr;
    }
}

// ---------------------------------------------------------------------------
// Pack x_dct[:, :L, :] (fp32) after the fp16 state region in d_out
// ---------------------------------------------------------------------------
__global__ __launch_bounds__(256)
void dct_copy_trunc(const float* __restrict__ xdct, void* __restrict__ dout,
                    const int* __restrict__ Lp)
{
    const int L = Lp[0];
    const size_t idx4 = (size_t)blockIdx.x * 256 + threadIdx.x;
    const size_t TC4 = (size_t)T_ * C_ / 4;
    const size_t b = idx4 / TC4;
    const size_t rem = idx4 % TC4;
    const int l  = (int)(rem / (C_ / 4));
    const int c4 = (int)(rem % (C_ / 4));
    if (l >= L) return;
    float4 val = ((const float4*)xdct)[idx4];
    float4* dst = (float4*)((char*)dout + (size_t)2 * B_ * L * C_);
    dst[((size_t)b * L + l) * (C_ / 4) + c4] = val;
}

// ---------------------------------------------------------------------------
extern "C" void kernel_launch(void* const* d_in, const int* in_sizes, int n_in,
                              void* d_out, int out_size, void* d_ws, size_t ws_size,
                              hipStream_t stream)
{
    (void)in_sizes; (void)n_in; (void)out_size; (void)ws_size;
    const float* x = (const float*)d_in[0];
    char* ws = (char*)d_ws;
    float*     meansum = (float*)(ws + WS_MEANSUM);
    float*     thr     = (float*)(ws + WS_THR);
    int*       Lp      = (int*)(ws + WS_L);
    _Float16*  Mf16    = (_Float16*)(ws + WS_MF16);
    _Float16*  MiT     = (_Float16*)(ws + WS_MIT);
    float*     xdct    = (float*)(ws + WS_XDCT);

    dct_init<<<1, 1024, 0, stream>>>(meansum, Lp, thr);
    dct_gen_M<<<(T_ * T_ + 255) / 256, 256, 0, stream>>>(Mf16);

    dim3 grid(C_ / 128, T_ / 64, B_);
    dct_gemm<0><<<grid, 256, 0, stream>>>(Mf16, x, xdct, nullptr, meansum, Lp);

    dct_stats<<<1, T_, 0, stream>>>(meansum, thr, Lp);
    dct_gen_MiT<<<(T_ * T_ + 255) / 256, 256, 0, stream>>>(MiT, Lp);

    dct_gemm<1><<<grid, 256, 0, stream>>>(MiT, xdct, nullptr, d_out, meansum, Lp);
    dct_copy_trunc<<<(B_ * T_ * C_ / 4 + 255) / 256, 256, 0, stream>>>(xdct, d_out, Lp);
}
